// BertEncoder_50620484551241
// MI455X (gfx1250) — matrix operand order repaired and checked by reading, not compiled, not run on hardware
//
#include <hip/hip_runtime.h>
#include <math.h>
#include <stdint.h>

// ---------------- problem constants (BERT-base, B=8, S=512) ----------------
#define B_   8
#define S_   512
#define H_   768
#define L_   12
#define NH_  12
#define F_   3072
#define D_   64
#define TOK  (B_ * S_)          // 4096 tokens

// ---------------- CDNA5 wave32 WMMA types ----------------
typedef __bf16 bf16_t;
typedef __attribute__((ext_vector_type(16))) __bf16 v16bf;
typedef __attribute__((ext_vector_type(8)))  __bf16 v8bf;
typedef __attribute__((ext_vector_type(8)))  float  v8f;
typedef __attribute__((__vector_size__(4 * sizeof(int)))) int v4i_t;

static __device__ __forceinline__ v8f wmma_bf16(v16bf a, v16bf b, v8f c) {
  // D = A(16x32 bf16) * B(32x16 bf16) + C(16x16 f32)
  return __builtin_amdgcn_wmma_f32_16x16x32_bf16(
      /*neg_a=*/false, a, /*neg_b=*/false, b,
      /*c_mod=*/(short)0, c, /*reuse_a=*/false, /*reuse_b=*/false);
}

// ---------------- async global->LDS copy (ASYNCcnt path) ----------------
static __device__ __forceinline__ void async_cp16(void* lds_dst, const void* gsrc) {
#if __has_builtin(__builtin_amdgcn_global_load_async_to_lds_b128)
  __builtin_amdgcn_global_load_async_to_lds_b128(
      (__attribute__((address_space(1))) v4i_t*)(uintptr_t)gsrc,
      (__attribute__((address_space(3))) v4i_t*)(uintptr_t)lds_dst, 0, 0);
#else
  asm volatile("global_load_async_to_lds_b128 %0, %1, off"
               :: "v"((uint32_t)(uintptr_t)lds_dst), "v"(gsrc) : "memory");
#endif
}
static __device__ __forceinline__ void async_wait0() {
#if __has_builtin(__builtin_amdgcn_s_wait_asynccnt)
  __builtin_amdgcn_s_wait_asynccnt(0);
#else
  asm volatile("s_wait_asynccnt 0x0" ::: "memory");
#endif
}

// A-fragment from row-major LDS tile X[m][k] (16x32), contiguous k order used here:
// lanes 0-15: M=lane, K 0..15
// lanes16-31: M=lane-16, K 16..31
static __device__ __forceinline__ v16bf load_frag_a(const bf16_t* X, int ldx,
                                                    int mBase, int kBase, int lane) {
  int m = lane & 15, half = lane >> 4;
  const bf16_t* p = X + (mBase + m) * ldx + kBase + 16 * half;
  v8bf lo = *(const v8bf*)p;
  v8bf hi = *(const v8bf*)(p + 8);
  v16bf a;
#pragma unroll
  for (int i = 0; i < 8; ++i) { a[i] = lo[i]; a[8 + i] = hi[i]; }
  return a;
}

// B-fragment when the tile is stored transposed in LDS: X[n][k].
// B layout: lanes 0-15: N=lane, K 0..15; lanes16-31: N=lane-16, K 16..31.
static __device__ __forceinline__ v16bf load_frag_b_tr(const bf16_t* X, int ldx,
                                                       int nBase, int kBase, int lane) {
  int n = lane & 15, khalf = lane >> 4;
  const bf16_t* p = X + (nBase + n) * ldx + kBase + 16 * khalf;
  v8bf lo = *(const v8bf*)p;
  v8bf hi = *(const v8bf*)(p + 8);
  v16bf b;
#pragma unroll
  for (int i = 0; i < 8; ++i) { b[i] = lo[i]; b[8 + i] = hi[i]; }
  return b;
}

// B-fragment from row-major LDS tile X[k][n] (element-wise; used for P*V only).
static __device__ __forceinline__ v16bf load_frag_b(const bf16_t* X, int ldx,
                                                    int kBase, int nBase, int lane) {
  int n = lane & 15, khalf = lane >> 4;
  const bf16_t* p = X + (kBase + khalf * 16) * ldx + nBase + n;
  v16bf b;
#pragma unroll
  for (int i = 0; i < 8; ++i) {
    b[2 * i]     = p[(2 * i) * ldx];
    b[2 * i + 1] = p[(2 * i + 1) * ldx];
  }
  return b;
}

static __device__ __forceinline__ float gelu_exact(float x) {
  return 0.5f * x * (1.0f + erff(x * 0.70710678118654752440f));
}

// =====================================================================
// One-time per launch: W[K][N] fp32  ->  Wt[N][K] bf16 (transposed cache).
// 32x32 tiles through LDS; both global sides coalesced.
// =====================================================================
__global__ __launch_bounds__(256) void transpose_cvt_kernel(
    const float* __restrict__ W, bf16_t* __restrict__ Wt, int K, int N) {
  __shared__ float tile[32][33];
  int k0 = blockIdx.y * 32, n0 = blockIdx.x * 32;
  int tr = threadIdx.x >> 3, tc = (threadIdx.x & 7) * 4;
  float4 f = *(const float4*)(W + (size_t)(k0 + tr) * N + n0 + tc);
  tile[tr][tc + 0] = f.x; tile[tr][tc + 1] = f.y;
  tile[tr][tc + 2] = f.z; tile[tr][tc + 3] = f.w;
  __syncthreads();
  bf16_t* d = Wt + (size_t)(n0 + tr) * K + k0 + tc;
#pragma unroll
  for (int i = 0; i < 4; ++i) d[i] = (bf16_t)tile[tc + i][tr];
}

// =====================================================================
// Embedding gather + LayerNorm -> fp32 h and bf16 hb.
// =====================================================================
__global__ __launch_bounds__(256) void embed_ln_kernel(
    const int* __restrict__ input_ids, const int* __restrict__ type_ids,
    const float* __restrict__ word_emb, const float* __restrict__ pos_emb,
    const float* __restrict__ type_emb, const float* __restrict__ scale,
    const float* __restrict__ bias, float* __restrict__ out,
    bf16_t* __restrict__ outb) {
  __shared__ float red[16];
  int row = blockIdx.x;
  int s   = row % S_;
  int tid = threadIdx.x, lane = tid & 31, w = tid >> 5;
  int id = input_ids[row], tp = type_ids[row];

  float v[3], sum = 0.f, sq = 0.f;
#pragma unroll
  for (int i = 0; i < 3; ++i) {
    int c = tid + i * 256;
    float x = word_emb[(size_t)id * H_ + c] + pos_emb[(size_t)s * H_ + c] +
              type_emb[(size_t)tp * H_ + c];
    v[i] = x; sum += x; sq += x * x;
  }
#pragma unroll
  for (int m = 16; m; m >>= 1) { sum += __shfl_xor(sum, m, 32); sq += __shfl_xor(sq, m, 32); }
  if (lane == 0) { red[w] = sum; red[8 + w] = sq; }
  __syncthreads();
  if (tid == 0) {
    float a = 0.f, b = 0.f;
#pragma unroll
    for (int i = 0; i < 8; ++i) { a += red[i]; b += red[8 + i]; }
    red[0] = a; red[8] = b;
  }
  __syncthreads();
  float mu  = red[0] * (1.0f / H_);
  float var = red[8] * (1.0f / H_) - mu * mu;
  float inv = rsqrtf(var + 1e-12f);
#pragma unroll
  for (int i = 0; i < 3; ++i) {
    int c = tid + i * 256;
    float y = (v[i] - mu) * inv * scale[c] + bias[c];
    out[(size_t)row * H_ + c]  = y;
    outb[(size_t)row * H_ + c] = (bf16_t)y;
  }
}

// =====================================================================
// bf16 WMMA GEMM: C[M,N] = act(A[M,K] @ W[K,N] + bias) (+ resid)
//  - A:  bf16 [M][K]   (activation cache)
//  - Wt: bf16 [N][K]   (pre-transposed weight cache)
//  - out: fp32 Cf and/or bf16 Cb (either may be null)
// Block tile 128x64, K-step 32, 8 waves x (2x2) WMMA tiles.
// LDS tiles double-buffered, filled by async global->LDS b128 copies.
// =====================================================================
#define BM   128
#define BN   64
#define BK   32
#define LDA  40   // BK + 8 pad: 16B-aligned rows, bank-conflict-free frag reads
#define LDBT 40

__global__ __launch_bounds__(256) void gemm_bias_act(
    const bf16_t* __restrict__ A, const bf16_t* __restrict__ Wt,
    const float* __restrict__ bias, const float* __restrict__ resid,
    float* __restrict__ Cf, bf16_t* __restrict__ Cb,
    int M, int N, int K, int act) {
  __shared__ bf16_t As[2][BM * LDA];
  __shared__ bf16_t Bs[2][BN * LDBT];

  int tid = threadIdx.x, lane = tid & 31, w = tid >> 5;
  int mBase = blockIdx.y * BM, nBase = blockIdx.x * BN;
  int wm = (w & 3) * 32;
  int wn = (w >> 2) * 32;

  v8f acc[2][2] = {};

  auto stage = [&](int buf, int k0) {
    // A tile: 128 x 32 bf16 (8KB) -> 2 x b128 per thread
    int ar = tid >> 1, ac = (tid & 1) * 16;
    const bf16_t* ga = A + (size_t)(mBase + ar) * K + k0 + ac;
    bf16_t* la = &As[buf][ar * LDA + ac];
    async_cp16(la,     ga);
    async_cp16(la + 8, ga + 8);
    // W tile (transposed): 64 x 32 bf16 (4KB) -> 1 x b128 per thread
    int br = tid >> 2, bc = (tid & 3) * 8;
    async_cp16(&Bs[buf][br * LDBT + bc],
               Wt + (size_t)(nBase + br) * K + k0 + bc);
  };

  stage(0, 0);
  int ib = 0;
  for (int k0 = 0; k0 < K; k0 += BK, ib ^= 1) {
    async_wait0();          // our async fills for buffer ib are done
    __syncthreads();        // everyone's fills done; prior reads retired
    if (k0 + BK < K) stage(ib ^ 1, k0 + BK);   // overlap next DMA with math

    const bf16_t* as = As[ib];
    const bf16_t* bs = Bs[ib];
    v16bf a0 = load_frag_a(as, LDA, wm + 0,  0, lane);
    v16bf a1 = load_frag_a(as, LDA, wm + 16, 0, lane);
    v16bf b0 = load_frag_b_tr(bs, LDBT, wn + 0,  0, lane);
    v16bf b1 = load_frag_b_tr(bs, LDBT, wn + 16, 0, lane);
    acc[0][0] = wmma_bf16(a0, b0, acc[0][0]);
    acc[0][1] = wmma_bf16(a0, b1, acc[0][1]);
    acc[1][0] = wmma_bf16(a1, b0, acc[1][0]);
    acc[1][1] = wmma_bf16(a1, b1, acc[1][1]);
  }

  // ---- epilogue: bias (+GELU) (+residual), fp32 and/or bf16 stores ----
#pragma unroll
  for (int mt = 0; mt < 2; ++mt) {
#pragma unroll
    for (int nt = 0; nt < 2; ++nt) {
      int col = nBase + wn + nt * 16 + (lane & 15);
      float bv = bias[col];
#pragma unroll
      for (int r = 0; r < 8; ++r) {
        int row = mBase + wm + mt * 16 + ((lane >> 4) << 3) + r;
        float val = acc[mt][nt][r] + bv;
        if (act == 1) val = gelu_exact(val);
        if (resid)    val += resid[(size_t)row * N + col];
        if (Cf) Cf[(size_t)row * N + col] = val;
        if (Cb) Cb[(size_t)row * N + col] = (bf16_t)val;
      }
    }
  }
}

// =====================================================================
// Attention: block = (batch*head, 64-query tile). Exact softmax with the
// full 64x512 fp32 score tile resident in the 320KB WGP LDS.
//   phase 1: S = (Q K^T)/8 + mask   (Q/K bf16, async-staged)
//   phase 2: row softmax -> P bf16 in LDS
//   phase 3: O = P V
// =====================================================================
#define TQ   64
#define LDH  72          // 64 + 8
#define LDP  (S_ + 8)    // 520

__global__ __launch_bounds__(256) void attention_kernel(
    const bf16_t* __restrict__ Qg, const bf16_t* __restrict__ Kg,
    const bf16_t* __restrict__ Vg, const int* __restrict__ ids,
    bf16_t* __restrict__ Og) {
  extern __shared__ __align__(16) char smem[];
  float*  scores = (float*)smem;                                   // TQ*S_ fp32
  bf16_t* P      = (bf16_t*)(smem + (size_t)TQ * S_ * 4);          // TQ*LDP
  bf16_t* Qs     = P + (size_t)TQ * LDP;                           // TQ*LDH
  bf16_t* KVs    = Qs + (size_t)TQ * LDH;                          // 64*LDH
  float*  maskv  = (float*)(KVs + (size_t)64 * LDH);               // S_

  int tid = threadIdx.x, lane = tid & 31, w = tid >> 5;
  int bh = blockIdx.x;
  int b = bh / NH_, head = bh % NH_;
  int qBase = blockIdx.y * TQ;
  size_t rowB = (size_t)b * S_;
  int colH = head * D_;

  int wm = (w >> 1) * 16;
  int wn = (w & 1) * 32;

  for (int i = tid; i < S_; i += 256)
    maskv[i] = (ids[rowB + i] == 0) ? -10000.0f : 0.0f;

  // async-stage Q tile (64x64 bf16, pure copy)
  int sr = tid >> 2, sc = (tid & 3) * 16;
  {
    const bf16_t* src = Qg + (rowB + qBase + sr) * H_ + colH + sc;
    bf16_t* dst = Qs + sr * LDH + sc;
    async_cp16(dst,     src);
    async_cp16(dst + 8, src + 8);
  }
  async_wait0();
  __syncthreads();

  // Q fragments are loop-invariant: hoist out of the key-tile loop
  v16bf qf0 = load_frag_a(Qs, LDH, wm, 0,  lane);
  v16bf qf1 = load_frag_a(Qs, LDH, wm, 32, lane);

  // -------- phase 1: scores --------
  for (int kt = 0; kt < S_ / 64; ++kt) {
    {
      const bf16_t* src = Kg + (rowB + kt * 64 + sr) * H_ + colH + sc;
      bf16_t* dst = KVs + sr * LDH + sc;
      async_cp16(dst,     src);
      async_cp16(dst + 8, src + 8);
    }
    async_wait0();
    __syncthreads();

    v8f acc0 = {}, acc1 = {};
    {
      v16bf b0 = load_frag_b_tr(KVs, LDH, wn + 0,  0,  lane);
      v16bf b1 = load_frag_b_tr(KVs, LDH, wn + 16, 0,  lane);
      acc0 = wmma_bf16(qf0, b0, acc0);
      acc1 = wmma_bf16(qf0, b1, acc1);
      b0 = load_frag_b_tr(KVs, LDH, wn + 0,  32, lane);
      b1 = load_frag_b_tr(KVs, LDH, wn + 16, 32, lane);
      acc0 = wmma_bf16(qf1, b0, acc0);
      acc1 = wmma_bf16(qf1, b1, acc1);
    }
    int colk = kt * 64 + wn + (lane & 15);
#pragma unroll
    for (int r = 0; r < 8; ++r) {
      int qr = wm + ((lane >> 4) << 3) + r;
      scores[qr * S_ + colk]      = acc0[r] * 0.125f + maskv[colk];
      scores[qr * S_ + colk + 16] = acc1[r] * 0.125f + maskv[colk + 16];
    }
    __syncthreads();
  }

  // -------- phase 2: exact softmax, P in bf16 --------
  for (int r8 = 0; r8 < 8; ++r8) {
    int qr = w * 8 + r8;
    const float* srow = scores + qr * S_;
    float mx = -1e30f;
    for (int i = lane; i < S_; i += 32) mx = fmaxf(mx, srow[i]);
#pragma unroll
    for (int m = 16; m; m >>= 1) mx = fmaxf(mx, __shfl_xor(mx, m, 32));
    float sum = 0.f;
    for (int i = lane; i < S_; i += 32) sum += __expf(srow[i] - mx);
#pragma unroll
    for (int m = 16; m; m >>= 1) sum += __shfl_xor(sum, m, 32);
    float inv = 1.0f / sum;
    bf16_t* prow = P + qr * LDP;
    for (int i = lane; i < S_; i += 32) prow[i] = (bf16_t)(__expf(srow[i] - mx) * inv);
  }
  __syncthreads();

  // -------- phase 3: O = P V --------
  v8f o0 = {}, o1 = {};
  for (int kt = 0; kt < S_ / 64; ++kt) {
    {
      const bf16_t* src = Vg + (rowB + kt * 64 + sr) * H_ + colH + sc;
      bf16_t* dst = KVs + sr * LDH + sc;
      async_cp16(dst,     src);
      async_cp16(dst + 8, src + 8);
    }
    async_wait0();
    __syncthreads();
#pragma unroll
    for (int ks = 0; ks < 2; ++ks) {
      v16bf a  = load_frag_a(P, LDP, wm, kt * 64 + ks * 32, lane);
      v16bf b0 = load_frag_b(KVs, LDH, ks * 32, wn + 0,  lane);
      v16bf b1 = load_frag_b(KVs, LDH, ks * 32, wn + 16, lane);
      o0 = wmma_bf16(a, b0, o0);
      o1 = wmma_bf16(a, b1, o1);
    }
    __syncthreads();
  }
  int cold = wn + (lane & 15);
#pragma unroll
  for (int r = 0; r < 8; ++r) {
    int qr = wm + ((lane >> 4) << 3) + r;
    size_t orow = (rowB + qBase + qr) * (size_t)H_ + colH;
    Og[orow + cold]      = (bf16_t)o0[r];
    Og[orow + cold + 16] = (bf16_t)o1[r];
  }
}

// =====================================================================
// LayerNorm over H=768 -> fp32 Yf and optional bf16 Yb.
// =====================================================================
__global__ __launch_bounds__(256) void layernorm_kernel(
    const float* __restrict__ X, const float* __restrict__ scale,
    const float* __restrict__ bias, float* __restrict__ Yf,
    bf16_t* __restrict__ Yb) {
  __shared__ float red[16];
  int row = blockIdx.x, tid = threadIdx.x, lane = tid & 31, w = tid >> 5;
  const float* x = X + (size_t)row * H_;
  float v[3], sum = 0.f, sq = 0.f;
#pragma unroll
  for (int i = 0; i < 3; ++i) {
    v[i] = x[tid + i * 256]; sum += v[i]; sq += v[i] * v[i];
  }
#pragma unroll
  for (int m = 16; m; m >>= 1) { sum += __shfl_xor(sum, m, 32); sq += __shfl_xor(sq, m, 32); }
  if (lane == 0) { red[w] = sum; red[8 + w] = sq; }
  __syncthreads();
  if (tid == 0) {
    float a = 0.f, b = 0.f;
#pragma unroll
    for (int i = 0; i < 8; ++i) { a += red[i]; b += red[8 + i]; }
    red[0] = a; red[8] = b;
  }
  __syncthreads();
  float mu  = red[0] * (1.0f / H_);
  float var = red[8] * (1.0f / H_) - mu * mu;
  float inv = rsqrtf(var + 1e-12f);
#pragma unroll
  for (int i = 0; i < 3; ++i) {
    int c = tid + i * 256;
    float y = (v[i] - mu) * inv * scale[c] + bias[c];
    Yf[(size_t)row * H_ + c] = y;
    if (Yb) Yb[(size_t)row * H_ + c] = (bf16_t)y;
  }
}

// =====================================================================
// Host-side orchestration
// =====================================================================
extern "C" void kernel_launch(void* const* d_in, const int* in_sizes, int n_in,
                              void* d_out, int out_size, void* d_ws, size_t ws_size,
                              hipStream_t stream) {
  (void)in_sizes; (void)n_in; (void)out_size; (void)ws_size;
  const int*   input_ids = (const int*)d_in[0];
  const int*   type_ids  = (const int*)d_in[1];
  const float* word_emb  = (const float*)d_in[2];
  const float* pos_emb   = (const float*)d_in[3];
  const float* type_emb  = (const float*)d_in[4];
  const float* emb_s     = (const float*)d_in[5];
  const float* emb_b     = (const float*)d_in[6];
  const float* wq = (const float*)d_in[7];  const float* bq = (const float*)d_in[8];
  const float* wk = (const float*)d_in[9];  const float* bk = (const float*)d_in[10];
  const float* wv = (const float*)d_in[11]; const float* bv = (const float*)d_in[12];
  const float* wo = (const float*)d_in[13]; const float* bo = (const float*)d_in[14];
  const float* l1s = (const float*)d_in[15]; const float* l1b = (const float*)d_in[16];
  const float* w1 = (const float*)d_in[17]; const float* b1 = (const float*)d_in[18];
  const float* w2 = (const float*)d_in[19]; const float* b2 = (const float*)d_in[20];
  const float* l2s = (const float*)d_in[21]; const float* l2b = (const float*)d_in[22];

  float* out = (float*)d_out;
  const size_t HID = (size_t)TOK * H_;
  const size_t WHH = (size_t)L_ * H_ * H_;
  const size_t WHF = (size_t)L_ * H_ * F_;

  // ---- workspace carve-up ----
  char* cur = (char*)d_ws;
  auto carve = [&](size_t bytes) {
    void* p = cur; cur += (bytes + 255) & ~(size_t)255; return p;
  };
  bf16_t* wqT = (bf16_t*)carve(WHH * 2);
  bf16_t* wkT = (bf16_t*)carve(WHH * 2);
  bf16_t* wvT = (bf16_t*)carve(WHH * 2);
  bf16_t* woT = (bf16_t*)carve(WHH * 2);
  bf16_t* w1T = (bf16_t*)carve(WHF * 2);
  bf16_t* w2T = (bf16_t*)carve(WHF * 2);
  float*  h   = (float*)carve(HID * 4);      // hidden (fp32, residual source)
  float*  t0f = (float*)carve(HID * 4);      // pre-LN1 temp
  float*  t1f = (float*)carve(HID * 4);      // LN1 out fp32 (FFN residual)
  float*  t2f = (float*)carve(HID * 4);      // pre-LN2 temp
  bf16_t* hb  = (bf16_t*)carve(HID * 2);     // bf16 hidden (QKV GEMM A)
  bf16_t* qb  = (bf16_t*)carve(HID * 2);
  bf16_t* kb  = (bf16_t*)carve(HID * 2);
  bf16_t* vb  = (bf16_t*)carve(HID * 2);
  bf16_t* cb  = (bf16_t*)carve(HID * 2);     // attention context
  bf16_t* t1b = (bf16_t*)carve(HID * 2);     // LN1 out bf16 (FFN1 A)
  bf16_t* fb  = (bf16_t*)carve((size_t)TOK * F_ * 2);  // FFN mid (bf16)

  const int smem_attn = TQ * S_ * 4 + TQ * LDP * 2 + TQ * LDH * 2
                      + 64 * LDH * 2 + S_ * 4;
  (void)hipFuncSetAttribute((const void*)attention_kernel,
                            hipFuncAttributeMaxDynamicSharedMemorySize, smem_attn);

  // ---- one-time (per launch) weight transpose + bf16 convert ----
  dim3 tHH(H_ / 32, H_ / 32);   // (24,24)
  dim3 tHF(F_ / 32, H_ / 32);   // (96,24)  w1: K=H,N=F
  dim3 tFH(H_ / 32, F_ / 32);   // (24,96)  w2: K=F,N=H
  for (int l = 0; l < L_; ++l) {
    transpose_cvt_kernel<<<tHH, 256, 0, stream>>>(wq + (size_t)l*H_*H_, wqT + (size_t)l*H_*H_, H_, H_);
    transpose_cvt_kernel<<<tHH, 256, 0, stream>>>(wk + (size_t)l*H_*H_, wkT + (size_t)l*H_*H_, H_, H_);
    transpose_cvt_kernel<<<tHH, 256, 0, stream>>>(wv + (size_t)l*H_*H_, wvT + (size_t)l*H_*H_, H_, H_);
    transpose_cvt_kernel<<<tHH, 256, 0, stream>>>(wo + (size_t)l*H_*H_, woT + (size_t)l*H_*H_, H_, H_);
    transpose_cvt_kernel<<<tHF, 256, 0, stream>>>(w1 + (size_t)l*H_*F_, w1T + (size_t)l*H_*F_, H_, F_);
    transpose_cvt_kernel<<<tFH, 256, 0, stream>>>(w2 + (size_t)l*H_*F_, w2T + (size_t)l*H_*F_, F_, H_);
  }

  embed_ln_kernel<<<TOK, 256, 0, stream>>>(input_ids, type_ids, word_emb, pos_emb,
                                           type_emb, emb_s, emb_b, h, hb);

  dim3 gProj(H_ / BN, TOK / BM);   // (12, 32)
  dim3 gFfn1(F_ / BN, TOK / BM);   // (48, 32)
  dim3 gAttn(B_ * NH_, S_ / TQ);   // (96, 8)

  for (int l = 0; l < L_; ++l) {
    bf16_t* wqTl = wqT + (size_t)l * H_ * H_;  const float* bql = bq + (size_t)l * H_;
    bf16_t* wkTl = wkT + (size_t)l * H_ * H_;  const float* bkl = bk + (size_t)l * H_;
    bf16_t* wvTl = wvT + (size_t)l * H_ * H_;  const float* bvl = bv + (size_t)l * H_;
    bf16_t* woTl = woT + (size_t)l * H_ * H_;  const float* bol = bo + (size_t)l * H_;
    bf16_t* w1Tl = w1T + (size_t)l * H_ * F_;  const float* b1l = b1 + (size_t)l * F_;
    bf16_t* w2Tl = w2T + (size_t)l * H_ * F_;  const float* b2l = b2 + (size_t)l * H_;

    gemm_bias_act<<<gProj, 256, 0, stream>>>(hb, wqTl, bql, nullptr, nullptr, qb, TOK, H_, H_, 0);
    gemm_bias_act<<<gProj, 256, 0, stream>>>(hb, wkTl, bkl, nullptr, nullptr, kb, TOK, H_, H_, 0);
    gemm_bias_act<<<gProj, 256, 0, stream>>>(hb, wvTl, bvl, nullptr, nullptr, vb, TOK, H_, H_, 0);

    attention_kernel<<<gAttn, 256, smem_attn, stream>>>(qb, kb, vb, input_ids, cb);

    // t0f = ctx @ wo + bo + h
    gemm_bias_act<<<gProj, 256, 0, stream>>>(cb, woTl, bol, h, t0f, nullptr, TOK, H_, H_, 0);
    layernorm_kernel<<<TOK, 256, 0, stream>>>(t0f, l1s + (size_t)l * H_, l1b + (size_t)l * H_,
                                              t1f, t1b);
    // fb = gelu(t1 @ w1 + b1)
    gemm_bias_act<<<gFfn1, 256, 0, stream>>>(t1b, w1Tl, b1l, nullptr, nullptr, fb, TOK, F_, H_, 1);
    // t2f = fb @ w2 + b2 + t1f
    gemm_bias_act<<<gProj, 256, 0, stream>>>(fb, w2Tl, b2l, t1f, t2f, nullptr, TOK, H_, F_, 0);

    bool last = (l == L_ - 1);
    layernorm_kernel<<<TOK, 256, 0, stream>>>(t2f, l2s + (size_t)l * H_, l2b + (size_t)l * H_,
                                              last ? out : h, last ? (bf16_t*)nullptr : hb);
  }
}
